// GPT_47167330845426
// MI455X (gfx1250) — matrix operand order repaired and checked by reading, not compiled, not run on hardware
//
#include <hip/hip_runtime.h>
#include <cstdint>
#include <cstddef>

// ---------------------------------------------------------------------------
// GPT-2-ish forward for MI455X (gfx1250, wave32, WMMA).
// All GEMMs: C = A[bf16, MxK row-major] * B^T[bf16, NxK row-major] via
// v_wmma_f32_16x16x32_bf16.  GEMM K-tiles are staged to LDS with
// global_load_async_to_lds_b128 (ASYNCcnt path) and double-buffered.
// Attention is flash-style (online softmax), one wave per 16-query tile.
// ---------------------------------------------------------------------------

typedef __attribute__((ext_vector_type(16))) __bf16 v16bf;
typedef __attribute__((ext_vector_type(8)))  __bf16 v8bf;
typedef __attribute__((ext_vector_type(8)))  float  v8f;

#define DEV static __device__ __forceinline__

namespace cfg {
constexpr int V  = 32000, D = 768, H = 12, L = 6, T = 1024, B = 4;
constexpr int HD = 64, FF = 3072;
constexpr int NT = B * T;                 // 4096 token rows
constexpr float SCALE = 0.125f;           // 1/sqrt(64)
constexpr float NEG   = -1e9f;
}

DEV __bf16 f2bf(float f) {                // round-to-nearest-even fp32->bf16
  union { float f; uint32_t u; } a; a.f = f;
  uint32_t r = (a.u + 0x7fffu + ((a.u >> 16) & 1u)) >> 16;
  union { uint16_t s; __bf16 b; } o; o.s = (uint16_t)r;
  return o.b;
}

DEV v8f wmma_bf16(v16bf a, v16bf b, v8f c) {
  return __builtin_amdgcn_wmma_f32_16x16x32_bf16(false, a, false, b,
                                                 (short)0, c, false, false);
}

DEV v16bf cat8(v8bf lo, v8bf hi) {
  return __builtin_shufflevector(lo, hi, 0,1,2,3,4,5,6,7,8,9,10,11,12,13,14,15);
}

// Async global->LDS copy, 16B per lane, tracked by ASYNCcnt (CDNA5 §10/§15.18).
// VDST = VGPR holding the wave-relative LDS byte address (low 32b of the flat
// shared-aperture pointer), VADDR = 64-bit global address.
DEV void cp_async_b128(uint32_t lds_addr, const void* gptr) {
  asm volatile("global_load_async_to_lds_b128 %0, %1, off"
               :: "v"(lds_addr), "v"((uint64_t)(uintptr_t)gptr) : "memory");
}
DEV void wait_async0() { asm volatile("s_wait_asynccnt 0x0" ::: "memory"); }
DEV void wait_async4() { asm volatile("s_wait_asynccnt 0x4" ::: "memory"); }

// ---------------------------------------------------------------------------
// Elementwise fp32 -> bf16 convert (for WTE: already [V,D] = [N,K] layout).
__global__ __launch_bounds__(256) void k_cvt(const float* __restrict__ src,
                                             __bf16* __restrict__ dst) {
  size_t i = (size_t)blockIdx.x * 256 + threadIdx.x;
  dst[i] = f2bf(src[i]);
}

// fp32 [K,N] -> bf16 [N,K] (transpose) so GEMM B-fragments are contiguous.
__global__ __launch_bounds__(256) void k_cvt_t(const float* __restrict__ W,
                                               __bf16* __restrict__ Wt,
                                               int K, int N) {
  size_t i = (size_t)blockIdx.x * 256 + threadIdx.x;   // i = n*K + k
  int n = (int)(i / K), k = (int)(i % K);
  Wt[i] = f2bf(W[(size_t)k * N + n]);
}

// x = WTE[idx] + WPE  (fp32 residual stream)
__global__ __launch_bounds__(256) void k_embed(const int* __restrict__ idx,
                                               const float* __restrict__ wte,
                                               const float* __restrict__ wpe,
                                               float* __restrict__ x) {
  size_t i = (size_t)blockIdx.x * 256 + threadIdx.x;
  int row = (int)(i / cfg::D), d = (int)(i % cfg::D);
  int t = row & (cfg::T - 1);
  x[i] = wte[(size_t)idx[row] * cfg::D + d] + wpe[(size_t)t * cfg::D + d];
}

// LayerNorm: fp32 in -> bf16 out (feeds the next GEMM's A operand).
__global__ __launch_bounds__(256) void k_ln(const float* __restrict__ x,
                                            const float* __restrict__ g,
                                            const float* __restrict__ b,
                                            __bf16* __restrict__ out) {
  __shared__ float red[256];
  const int tid = threadIdx.x, row = blockIdx.x;
  const float* xr = x + (size_t)row * cfg::D;
  float v0 = xr[tid], v1 = xr[tid + 256], v2 = xr[tid + 512];

  red[tid] = v0 + v1 + v2; __syncthreads();
  for (int off = 128; off > 0; off >>= 1) {
    if (tid < off) red[tid] += red[tid + off];
    __syncthreads();
  }
  float mean = red[0] * (1.0f / cfg::D);
  __syncthreads();
  float d0 = v0 - mean, d1 = v1 - mean, d2 = v2 - mean;
  red[tid] = d0*d0 + d1*d1 + d2*d2; __syncthreads();
  for (int off = 128; off > 0; off >>= 1) {
    if (tid < off) red[tid] += red[tid + off];
    __syncthreads();
  }
  float rstd = rsqrtf(red[0] * (1.0f / cfg::D) + 1e-5f);

  __bf16* orow = out + (size_t)row * cfg::D;
  orow[tid      ] = f2bf(d0 * rstd * g[tid      ] + b[tid      ]);
  orow[tid + 256] = f2bf(d1 * rstd * g[tid + 256] + b[tid + 256]);
  orow[tid + 512] = f2bf(d2 * rstd * g[tid + 512] + b[tid + 512]);
}

// ---------------------------------------------------------------------------
// WMMA GEMM: C[M,N] = A[M,K] * Bt[N,K]^T, fused epilogues.
// Block = 256 thr (8 waves), tile 128x128, BK=32, wave tile 64x32.
// LDS double-buffered; tiles staged via async global->LDS b128 copies.
enum { EPI_BF16 = 0, EPI_RES_F32 = 1, EPI_GELU_BF16 = 2,
       EPI_BIAS_RES_F32 = 3, EPI_F32 = 4 };

template <int EPI>
__global__ __launch_bounds__(256) void k_gemm(
    const __bf16* __restrict__ A, const __bf16* __restrict__ Bt,
    int M, int N, int K,
    const float* __restrict__ bias, const float* __restrict__ res,
    float* __restrict__ outF, __bf16* __restrict__ outB) {
  constexpr int BM = 128, BN = 128, BK = 32;
  __shared__ __bf16 As[2][BM][BK];
  __shared__ __bf16 Bs[2][BN][BK];

  const int tid  = threadIdx.x;
  const int lane = tid & 31, wave = tid >> 5;
  const int l15  = lane & 15, lh = lane >> 4, kb = lh * 8;
  const int wm   = wave >> 2, wn = wave & 3;              // 2 x 4 wave grid
  const int m0   = blockIdx.y * BM, n0 = blockIdx.x * BN;

  v8f acc[4][2] = {};                                     // 64 x 32 per wave

  // Stage one K-tile into buffer p: 4 async b128 copies per wave
  // (each thread: 2 x 16B of A, 2 x 16B of B).
  auto issue_tile = [&](int p, int kt) {
#pragma unroll
    for (int i = 0; i < 2; ++i) {
      int c = tid + i * 256;                              // 0..511
      int row = c >> 2, ch = (c & 3) * 8;
      cp_async_b128((uint32_t)(uintptr_t)&As[p][row][ch],
                    A  + (size_t)(m0 + row) * K + kt + ch);
      cp_async_b128((uint32_t)(uintptr_t)&Bs[p][row][ch],
                    Bt + (size_t)(n0 + row) * K + kt + ch);
    }
  };

  const int ntile = K / BK;
  issue_tile(0, 0);
  for (int it = 0; it < ntile; ++it) {
    const int p = it & 1;
    if (it + 1 < ntile) {
      issue_tile(p ^ 1, (it + 1) * BK);   // prefetch next tile (in-flight)
      wait_async4();                      // oldest 4 (tile `it`) complete
    } else {
      wait_async0();
    }
    __syncthreads();                      // whole block's tile `it` in LDS

    v16bf af[4], bfr[2];
#pragma unroll
    for (int i = 0; i < 4; ++i) {
      int r = wm * 64 + i * 16 + l15;                     // A: lane = row m
      af[i] = cat8(*(const v8bf*)&As[p][r][kb],
                   *(const v8bf*)&As[p][r][kb + 16]);
    }
#pragma unroll
    for (int j = 0; j < 2; ++j) {
      int r = wn * 32 + j * 16 + l15;                     // B: lane = col n
      bfr[j] = cat8(*(const v8bf*)&Bs[p][r][kb],
                    *(const v8bf*)&Bs[p][r][kb + 16]);    // k table order
    }
#pragma unroll
    for (int i = 0; i < 4; ++i)
#pragma unroll
      for (int j = 0; j < 2; ++j)
        acc[i][j] = wmma_bf16(af[i], bfr[j], acc[i][j]);
    __syncthreads();                      // all reads of buf p done
  }

  // ---- epilogue: C layout lane=n, vgpr r -> m = r + 8*(lane>=16) ----
#pragma unroll
  for (int i = 0; i < 4; ++i)
#pragma unroll
    for (int j = 0; j < 2; ++j)
#pragma unroll
      for (int r = 0; r < 8; ++r) {
        int m = m0 + wm * 64 + i * 16 + r + 8 * lh;
        int n = n0 + wn * 32 + j * 16 + l15;
        size_t o = (size_t)m * N + n;
        float v = acc[i][j][r];
        if constexpr (EPI == EPI_GELU_BF16 || EPI == EPI_BIAS_RES_F32) v += bias[n];
        if constexpr (EPI == EPI_GELU_BF16)
          v = 0.5f * v * (1.0f + erff(v * 0.70710678118654752f));
        if constexpr (EPI == EPI_RES_F32 || EPI == EPI_BIAS_RES_F32) v += res[o];
        if constexpr (EPI == EPI_BF16 || EPI == EPI_GELU_BF16) outB[o] = f2bf(v);
        else                                                   outF[o] = v;
      }
}

// ---------------------------------------------------------------------------
// V transpose: qkv[:, 1536 + h*64 + d] -> Vt[(b,h,d), t]  (B-operand layout)
__global__ __launch_bounds__(256) void k_vt(const __bf16* __restrict__ qkv,
                                            __bf16* __restrict__ vt) {
  size_t i = (size_t)blockIdx.x * 256 + threadIdx.x;
  int row = (int)(i / cfg::D), col = (int)(i % cfg::D);
  int h = col >> 6, d = col & 63;
  int b = row >> 10, t = row & (cfg::T - 1);
  vt[((size_t)((b * cfg::H + h) * cfg::HD + d)) * cfg::T + t] =
      qkv[(size_t)row * (3 * cfg::D) + 2 * cfg::D + col];
}

// Flash attention: one wave per (b, h, 16-query tile).  Scores + AV via WMMA.
__global__ __launch_bounds__(256) void k_attn(const __bf16* __restrict__ qkv,
                                              const __bf16* __restrict__ vt,
                                              __bf16* __restrict__ y) {
  __shared__ __bf16 Psh[8][16 * 32];                      // per-wave P tile
  const int lane = threadIdx.x & 31, wv = threadIdx.x >> 5;
  const int l15 = lane & 15, lh = lane >> 4, kb = lh * 8;
  const int gw = blockIdx.x * 8 + wv;                     // 0..3071
  const int qt = gw & 63, bh = gw >> 6;
  const int h = bh % cfg::H, b = bh / cfg::H;
  const int q0 = qt * 16;
  const int QKVW = 3 * cfg::D;

  // Q fragments (A layout): row = q0 + l15, d split into 2 chunks of 32.
  const __bf16* qrow = qkv + (size_t)(b * cfg::T + q0 + l15) * QKVW + h * cfg::HD;
  v16bf qf[2];
#pragma unroll
  for (int c = 0; c < 2; ++c)
    qf[c] = cat8(*(const v8bf*)(qrow + c * 32 + kb),
                 *(const v8bf*)(qrow + c * 32 + kb + 16));

  v8f o[4] = {};
  float mrow[8], lrow[8];
#pragma unroll
  for (int r = 0; r < 8; ++r) { mrow[r] = -1e30f; lrow[r] = 0.0f; }

  const int nk = (q0 + 47) >> 5;                          // 32-key chunks
  for (int kc = 0; kc < nk; ++kc) {
    const int k0 = kc * 32;
    v8f s0 = {}, s1 = {};
#pragma unroll
    for (int c = 0; c < 2; ++c) {                         // d-chunks of 32
      const __bf16* k0r = qkv + (size_t)(b * cfg::T + k0 + l15) * QKVW
                              + cfg::D + h * cfg::HD + c * 32;
      const __bf16* k1r = k0r + (size_t)16 * QKVW;
      s0 = wmma_bf16(qf[c], cat8(*(const v8bf*)(k0r + kb), *(const v8bf*)(k0r + kb + 16)), s0);
      s1 = wmma_bf16(qf[c], cat8(*(const v8bf*)(k1r + kb), *(const v8bf*)(k1r + kb + 16)), s1);
    }
    const bool domask = (k0 + 31 > q0);
#pragma unroll
    for (int r = 0; r < 8; ++r) {
      int row = r + 8 * lh;
      float a  = s0[r] * cfg::SCALE;
      float bb = s1[r] * cfg::SCALE;
      if (domask) {
        if (k0 + l15      > q0 + row) a  = cfg::NEG;
        if (k0 + 16 + l15 > q0 + row) bb = cfg::NEG;
      }
      float mx = fmaxf(a, bb);
      mx = fmaxf(mx, __shfl_xor(mx, 1, 32));
      mx = fmaxf(mx, __shfl_xor(mx, 2, 32));
      mx = fmaxf(mx, __shfl_xor(mx, 4, 32));
      mx = fmaxf(mx, __shfl_xor(mx, 8, 32));              // row max (16 lanes)
      float mnew  = fmaxf(mrow[r], mx);
      float alpha = __expf(mrow[r] - mnew);
      float pa = __expf(a - mnew), pb = __expf(bb - mnew);
      float ps = pa + pb;
      ps += __shfl_xor(ps, 1, 32);
      ps += __shfl_xor(ps, 2, 32);
      ps += __shfl_xor(ps, 4, 32);
      ps += __shfl_xor(ps, 8, 32);                        // row sum
      lrow[r] = lrow[r] * alpha + ps;
      mrow[r] = mnew;
#pragma unroll
      for (int j = 0; j < 4; ++j) o[j][r] *= alpha;       // rescale prior acc
      Psh[wv][row * 32 + l15]      = f2bf(pa);            // C-layout -> LDS
      Psh[wv][row * 32 + 16 + l15] = f2bf(pb);
    }
    // P as A-fragment (row-major 16x32 in LDS, wave-internal DS ordering).
    const __bf16* prow = &Psh[wv][l15 * 32];
    v16bf pf = cat8(*(const v8bf*)(prow + kb), *(const v8bf*)(prow + kb + 16));
#pragma unroll
    for (int j = 0; j < 4; ++j) {                         // output d-tiles
      const __bf16* vr = vt + (size_t)((b * cfg::H + h) * cfg::HD + j * 16 + l15)
                            * cfg::T + k0;
      o[j] = wmma_bf16(pf, cat8(*(const v8bf*)(vr + kb), *(const v8bf*)(vr + kb + 16)), o[j]);
    }
  }

  float inv[8];
#pragma unroll
  for (int r = 0; r < 8; ++r) inv[r] = 1.0f / lrow[r];
#pragma unroll
  for (int j = 0; j < 4; ++j)
#pragma unroll
    for (int r = 0; r < 8; ++r) {
      int m = q0 + r + 8 * lh;
      y[(size_t)(b * cfg::T + m) * cfg::D + h * cfg::HD + j * 16 + l15] =
          f2bf(o[j][r] * inv[r]);
    }
}

// ---------------------------------------------------------------------------
extern "C" void kernel_launch(void* const* d_in, const int* in_sizes, int n_in,
                              void* d_out, int out_size, void* d_ws, size_t ws_size,
                              hipStream_t stream) {
  using namespace cfg;
  (void)in_sizes; (void)n_in; (void)out_size; (void)ws_size;

  const int*   idx   = (const int*)  d_in[0];
  const float* WTE   = (const float*)d_in[1];
  const float* WPE   = (const float*)d_in[2];
  const float* ln1_g = (const float*)d_in[3];
  const float* ln1_b = (const float*)d_in[4];
  const float* Wqkv  = (const float*)d_in[5];
  const float* Wproj = (const float*)d_in[6];
  const float* ln2_g = (const float*)d_in[7];
  const float* ln2_b = (const float*)d_in[8];
  const float* W1    = (const float*)d_in[9];
  const float* b1    = (const float*)d_in[10];
  const float* W2    = (const float*)d_in[11];
  const float* b2    = (const float*)d_in[12];
  const float* lnf_g = (const float*)d_in[13];
  const float* lnf_b = (const float*)d_in[14];
  float* logits = (float*)d_out;

  // workspace layout (bytes)
  char* ws = (char*)d_ws;
  constexpr size_t OFF_X      = 0;                              // fp32 [4096,768]
  constexpr size_t OFF_HBF    = OFF_X      + (size_t)NT * D * 4;
  constexpr size_t OFF_QKV    = OFF_HBF    + (size_t)NT * D * 2;
  constexpr size_t OFF_VT     = OFF_QKV    + (size_t)NT * 3 * D * 2;
  constexpr size_t OFF_Y      = OFF_VT     + (size_t)NT * D * 2;
  constexpr size_t OFF_FF     = OFF_Y      + (size_t)NT * D * 2;
  constexpr size_t OFF_WQKVT  = OFF_FF     + (size_t)NT * FF * 2;
  constexpr size_t OFF_WPROJT = OFF_WQKVT  + (size_t)3 * D * D * 2;
  constexpr size_t OFF_W1T    = OFF_WPROJT + (size_t)D * D * 2;
  constexpr size_t OFF_W2T    = OFF_W1T    + (size_t)D * FF * 2;
  constexpr size_t OFF_WTEB   = OFF_W2T    + (size_t)FF * D * 2;

  float*  x      = (float*) (ws + OFF_X);
  __bf16* hbf    = (__bf16*)(ws + OFF_HBF);
  __bf16* qkvb   = (__bf16*)(ws + OFF_QKV);
  __bf16* vtb    = (__bf16*)(ws + OFF_VT);
  __bf16* ybf    = (__bf16*)(ws + OFF_Y);
  __bf16* ffb    = (__bf16*)(ws + OFF_FF);
  __bf16* WqkvT  = (__bf16*)(ws + OFF_WQKVT);
  __bf16* WprojT = (__bf16*)(ws + OFF_WPROJT);
  __bf16* W1T    = (__bf16*)(ws + OFF_W1T);
  __bf16* W2T    = (__bf16*)(ws + OFF_W2T);
  __bf16* WTEb   = (__bf16*)(ws + OFF_WTEB);

  // WTE is [V,D] = [N,K]: straight convert (LM head B operand).
  k_cvt<<<(V * D) / 256, 256, 0, stream>>>(WTE, WTEb);
  k_embed<<<(NT * D) / 256, 256, 0, stream>>>(idx, WTE, WPE, x);

  const dim3 blk(256);
  for (int l = 0; l < L; ++l) {
    const float* wqkv = Wqkv  + (size_t)l * D * 3 * D;
    const float* wp   = Wproj + (size_t)l * D * D;
    const float* w1   = W1    + (size_t)l * D * FF;
    const float* w2   = W2    + (size_t)l * FF * D;

    k_cvt_t<<<(3 * D * D) / 256, blk, 0, stream>>>(wqkv, WqkvT, D, 3 * D);
    k_cvt_t<<<(D * D) / 256,     blk, 0, stream>>>(wp,   WprojT, D, D);
    k_cvt_t<<<(D * FF) / 256,    blk, 0, stream>>>(w1,   W1T,    D, FF);
    k_cvt_t<<<(FF * D) / 256,    blk, 0, stream>>>(w2,   W2T,    FF, D);

    k_ln<<<NT, blk, 0, stream>>>(x, ln1_g + l * D, ln1_b + l * D, hbf);

    k_gemm<EPI_BF16><<<dim3((3 * D) / 128, NT / 128), blk, 0, stream>>>(
        hbf, WqkvT, NT, 3 * D, D, nullptr, nullptr, nullptr, qkvb);

    k_vt<<<(NT * D) / 256, blk, 0, stream>>>(qkvb, vtb);
    k_attn<<<(B * H * (T / 16)) / 8, blk, 0, stream>>>(qkvb, vtb, ybf);

    k_gemm<EPI_RES_F32><<<dim3(D / 128, NT / 128), blk, 0, stream>>>(
        ybf, WprojT, NT, D, D, nullptr, x, x, nullptr);

    k_ln<<<NT, blk, 0, stream>>>(x, ln2_g + l * D, ln2_b + l * D, hbf);

    k_gemm<EPI_GELU_BF16><<<dim3(FF / 128, NT / 128), blk, 0, stream>>>(
        hbf, W1T, NT, FF, D, b1 + (size_t)l * FF, nullptr, nullptr, ffb);

    k_gemm<EPI_BIAS_RES_F32><<<dim3(D / 128, NT / 128), blk, 0, stream>>>(
        ffb, W2T, NT, D, FF, b2 + (size_t)l * D, x, x, nullptr);
  }

  k_ln<<<NT, blk, 0, stream>>>(x, lnf_g, lnf_b, hbf);

  // tied LM head: logits[4096, 32000] = xf @ WTE^T  (WTEb already [N,K])
  k_gemm<EPI_F32><<<dim3(V / 128, NT / 128), blk, 0, stream>>>(
      hbf, WTEb, NT, V, D, nullptr, nullptr, logits, nullptr);
}
